// VRAE_19189913878726
// MI455X (gfx1250) — compile-verified, matrix operand order checked by reading, not run on hardware
//
#include <hip/hip_runtime.h>
#include <hip/hip_bf16.h>
#include <math.h>

// ---------------------------------------------------------------------------
// VRAE for MI455X (gfx1250, wave32, WMMA + async global->LDS staging).
//   B=512, T=512, D=128, H=1024, L=128
//  - bf16 inputs / fp32 accumulation via v_wmma_f32_16x16x32_bf16.
//  - Encoder step: h' = tanh([x_t | h] @ [W_enc_in ; W_enc_rec] + b_enc).
//  - Decoder step is affine in h: precompute W_comb = W_out@W_dec_in + W_dec_rec,
//    b_comb = b_out@W_dec_in + b_dec; each step is ONE GEMM h @ [W_comb | W_out].
//  - Weights pre-packed into WMMA B-fragment layout once per call; per K-step
//    the 4KB of B fragments are DMA'd to LDS once per workgroup with
//    global_load_async_to_lds_b128, double buffered (ASYNCcnt pipeline).
//  - sched_barrier keeps the 8 ds_load_b128 issued as one clause followed by
//    4 back-to-back WMMAs (single s_wait_dscnt per K-step).
// ---------------------------------------------------------------------------

#define Bc 512
#define Tc 512
#define Dc 128
#define Hc 1024
#define Lc 128

typedef unsigned short u16t;
typedef __attribute__((ext_vector_type(16))) __bf16 v16bf;
typedef __attribute__((ext_vector_type(8)))  float  v8f;

// ---- bf16 helpers (bit-level, round-to-nearest-even) ----------------------
__device__ __forceinline__ u16t f2bf(float f) {
    unsigned int x = __float_as_uint(f);
    unsigned int r = x + 0x7FFFu + ((x >> 16) & 1u);
    return (u16t)(r >> 16);
}
__device__ __forceinline__ float bf2f(u16t u) {
    return __uint_as_float(((unsigned int)u) << 16);
}

__device__ __forceinline__ void sched_fence() {
#if __has_builtin(__builtin_amdgcn_sched_barrier)
    __builtin_amdgcn_sched_barrier(0);
#endif
}

// ---- async global->LDS (CDNA5, ASYNCcnt) ----------------------------------
// 16 bytes per lane; lds byte address = low 32 bits of generic shared pointer.
#define ASYNC_LOAD_B128(laddr, gptr)                                          \
    asm volatile("global_load_async_to_lds_b128 %0, %1, off" ::"v"(laddr),    \
                 "v"(gptr)                                                    \
                 : "memory")

__device__ __forceinline__ void wait_async_le2() {
#if __has_builtin(__builtin_amdgcn_s_wait_asynccnt)
    __builtin_amdgcn_s_wait_asynccnt(2);
#else
    asm volatile("s_wait_asynccnt 0x2" ::: "memory");
#endif
}
__device__ __forceinline__ void wait_async_le0() {
#if __has_builtin(__builtin_amdgcn_s_wait_asynccnt)
    __builtin_amdgcn_s_wait_asynccnt(0);
#else
    asm volatile("s_wait_asynccnt 0x0" ::: "memory");
#endif
}

// ---- WMMA A-fragment load (16x32 bf16, ISA 7.12.2 layout) -----------------
// lane: m = lane&15, khalf = lane>>4. VGPR p holds bf16 pair (k0,k0+1):
//   p<4: k0 = 2p + 8*khalf ;  p>=4: k0 = 16 + 2(p-4) + 8*khalf
// (pairs are contiguous -> compiler merges into two b128 loads)
__device__ __forceinline__ v16bf load_a(const u16t* __restrict__ row,
                                        int kbase, int khalf) {
    union { v16bf bf; unsigned int d[8]; } f;
#pragma unroll
    for (int p = 0; p < 8; ++p) {
        int k0 = (p < 4 ? 2 * p : 16 + 2 * (p - 4)) + 8 * khalf + kbase;
        f.d[p] = *reinterpret_cast<const unsigned int*>(row + k0);
    }
    return f.bf;
}

// ---- WMMA B-fragment read from LDS (32B/lane, two ds_load_b128) -----------
__device__ __forceinline__ v16bf load_b_lds(const u16t* sm, int frag,
                                            int lane) {
    const uint4* q =
        reinterpret_cast<const uint4*>(sm + frag * 512 + lane * 16);
    union { v16bf bf; uint4 d[2]; } f;
    f.d[0] = q[0];
    f.d[1] = q[1];
    return f.bf;
}

// ===========================================================================
// Pre/post-processing kernels
// ===========================================================================

__global__ void cvt_bf16_kernel(const float* __restrict__ src,
                                u16t* __restrict__ dst, int n) {
    int i = blockIdx.x * blockDim.x + threadIdx.x;
    if (i < n) dst[i] = f2bf(src[i]);
}

__global__ void zero_u16_kernel(u16t* __restrict__ dst, int n) {
    int i = blockIdx.x * blockDim.x + threadIdx.x;
    if (i < n) dst[i] = 0;  // 0x0000 == +0.0 bf16
}

// Pack fp32 row-major weight [K x N] (row stride ldn) into WMMA B fragments.
// Fragment (kt, nt) -> dst[((kt+ktOff)*dstNtiles + (nt+ntOff))*512 + lane*16 + e]
// B layout mirrors A with n = lane&15: e -> (p=e/2, j=e&1),
//   k = kt*32 + (p<4 ? 2p : 16+2(p-4)) + 8*(lane>>4) + j
__global__ void pack_b_kernel(const float* __restrict__ src, int K, int N,
                              int ldn, u16t* __restrict__ dst, int dstNtiles,
                              int ktOff, int ntOff) {
    int tid = blockIdx.x * blockDim.x + threadIdx.x;
    int total = (K / 32) * (N / 16) * 512;
    if (tid >= total) return;
    int e    = tid & 15;
    int lane = (tid >> 4) & 31;
    int frag = tid >> 9;
    int ntiles = N / 16;
    int nt = frag % ntiles;
    int kt = frag / ntiles;
    int n = nt * 16 + (lane & 15);
    int khalf = lane >> 4;
    int p = e >> 1, j = e & 1;
    int k = kt * 32 + (p < 4 ? 2 * p : 16 + 2 * (p - 4)) + 8 * khalf + j;
    size_t didx = (((size_t)(kt + ktOff) * dstNtiles + (nt + ntOff)) * 32 +
                   lane) * 16 + e;
    dst[didx] = f2bf(src[(size_t)k * ldn + n]);
}

// W_comb[i][j] = W_dec_rec[i][j] + sum_d W_out[i][d] * W_dec_in[d][j]
__global__ void combine_w_kernel(const float* __restrict__ W_out,
                                 const float* __restrict__ W_dec_in,
                                 const float* __restrict__ W_dec_rec,
                                 float* __restrict__ W_comb) {
    int tid = blockIdx.x * blockDim.x + threadIdx.x;
    if (tid >= Hc * Hc) return;
    int i = tid / Hc, j = tid % Hc;
    float acc = W_dec_rec[(size_t)i * Hc + j];
#pragma unroll 4
    for (int d = 0; d < Dc; ++d)
        acc += W_out[(size_t)i * Dc + d] * W_dec_in[(size_t)d * Hc + j];
    W_comb[(size_t)i * Hc + j] = acc;
}

// b_comb[j] = b_dec[j] + sum_d b_out[d] * W_dec_in[d][j]
__global__ void combine_b_kernel(const float* __restrict__ b_out,
                                 const float* __restrict__ W_dec_in,
                                 const float* __restrict__ b_dec,
                                 float* __restrict__ b_comb) {
    int j = blockIdx.x * blockDim.x + threadIdx.x;
    if (j >= Hc) return;
    float acc = b_dec[j];
#pragma unroll 4
    for (int d = 0; d < Dc; ++d)
        acc += b_out[d] * W_dec_in[(size_t)d * Hc + j];
    b_comb[j] = acc;
}

// mu = h@W_mu + b_mu ; std = 1e-6 + softplus(h@W_std + b_std) ;
// z = mu + std*eps. Writes mu/std into d_out tail, z to workspace.
__global__ void latent_kernel(const u16t* __restrict__ h,
                              const float* __restrict__ W_mu,
                              const float* __restrict__ b_mu,
                              const float* __restrict__ W_std,
                              const float* __restrict__ b_std,
                              const float* __restrict__ eps,
                              float* __restrict__ out, float* __restrict__ z) {
    int tid = blockIdx.x * blockDim.x + threadIdx.x;
    if (tid >= Bc * Lc) return;
    int b = tid / Lc, l = tid % Lc;
    float am = b_mu[l], as = b_std[l];
    const u16t* hr = h + (size_t)b * Hc;
    for (int k = 0; k < Hc; ++k) {
        float hv = bf2f(hr[k]);
        am += hv * W_mu[(size_t)k * Lc + l];
        as += hv * W_std[(size_t)k * Lc + l];
    }
    float sp = (as > 20.0f) ? as : log1pf(expf(as));
    float sd = 1e-6f + sp;
    size_t ylen = (size_t)Bc * Tc * Dc;
    out[ylen + tid] = am;                    // mu_enc
    out[ylen + (size_t)Bc * Lc + tid] = sd;  // stddev_enc
    z[tid] = am + sd * eps[tid];
}

// h_dec0 = tanh(z @ W_h0 + b_h0)  -> bf16
__global__ void h0_kernel(const float* __restrict__ z,
                          const float* __restrict__ W_h0,
                          const float* __restrict__ b_h0,
                          u16t* __restrict__ hdec) {
    int tid = blockIdx.x * blockDim.x + threadIdx.x;
    if (tid >= Bc * Hc) return;
    int b = tid / Hc, j = tid % Hc;
    float acc = b_h0[j];
#pragma unroll 4
    for (int k = 0; k < Lc; ++k)
        acc += z[(size_t)b * Lc + k] * W_h0[(size_t)k * Hc + j];
    hdec[tid] = f2bf(tanhf(acc));
}

// ===========================================================================
// Recurrent WMMA GEMM kernels (block=128 threads=4 waves, WG tile 64x64,
// wave tile 16x64; B fragments double-buffered in LDS via async DMA)
// ===========================================================================

// Encoder step: hout = tanh([x_t | hin] @ Wpk(K=1152,Ntiles=64) + b_enc)
__global__ __launch_bounds__(128)
void enc_step_kernel(const u16t* __restrict__ xbf,   // [B*T*D] bf16
                     const u16t* __restrict__ hin,   // [B*H]  bf16
                     const u16t* __restrict__ wpk,   // packed frags
                     const float* __restrict__ bias, // b_enc[H]
                     u16t* __restrict__ hout, int t) {
    __shared__ u16t smem[2 * 2048];  // 2 bufs x 4 frags x 512 u16 = 8KB
    const int tid   = threadIdx.x;
    const int lane  = tid & 31;
    const int wave  = tid >> 5;
    const int khalf = lane >> 4;
    const int row   = blockIdx.x * 64 + wave * 16 + (lane & 15);  // batch idx
    const int nt0   = blockIdx.y * 4;

    const u16t* arow_x = xbf + ((size_t)row * Tc + t) * Dc;
    const u16t* arow_h = hin + (size_t)row * Hc;

    // stage K-step s's 4 B-fragments (4KB) into LDS buffer (s&1)
    auto stage = [&](int s) {
        const u16t* g = wpk + ((size_t)s * 64 + nt0) * 512 + (size_t)tid * 8;
        unsigned l0 =
            (unsigned)(size_t)(smem + (size_t)(s & 1) * 2048 + tid * 8);
        ASYNC_LOAD_B128(l0, g);
        ASYNC_LOAD_B128(l0 + 2048u, g + 1024);
    };

    v8f acc[4] = {};
    // 8 ds_load_b128 issued as one group, then 4 back-to-back WMMAs
    auto mm4 = [&](v16bf a, int s) {
        const u16t* sb = smem + (size_t)(s & 1) * 2048;
        v16bf b0 = load_b_lds(sb, 0, lane);
        v16bf b1 = load_b_lds(sb, 1, lane);
        v16bf b2 = load_b_lds(sb, 2, lane);
        v16bf b3 = load_b_lds(sb, 3, lane);
        sched_fence();  // keep all ds loads before the WMMA group
        acc[0] = __builtin_amdgcn_wmma_f32_16x16x32_bf16(
            false, a, false, b0, (short)0, acc[0], false, false);
        acc[1] = __builtin_amdgcn_wmma_f32_16x16x32_bf16(
            false, a, false, b1, (short)0, acc[1], false, false);
        acc[2] = __builtin_amdgcn_wmma_f32_16x16x32_bf16(
            false, a, false, b2, (short)0, acc[2], false, false);
        acc[3] = __builtin_amdgcn_wmma_f32_16x16x32_bf16(
            false, a, false, b3, (short)0, acc[3], false, false);
    };

    stage(0);
    // ---- x-projection K tiles (kt 0..3) ----
#pragma unroll 1
    for (int s = 0; s < 4; ++s) {
        stage(s + 1);
        v16bf a = load_a(arow_x, s * 32, khalf);  // overlap with barrier wait
        wait_async_le2();
        __syncthreads();
        mm4(a, s);
        __syncthreads();
    }
    // ---- recurrent K tiles (kt 4..34, pipelined) ----
#pragma unroll 1
    for (int s = 4; s < 35; ++s) {
        stage(s + 1);
        v16bf a = load_a(arow_h, (s - 4) * 32, khalf);
        wait_async_le2();
        __syncthreads();
        mm4(a, s);
        __syncthreads();
    }
    // ---- last K tile (kt 35) ----
    {
        v16bf a = load_a(arow_h, 31 * 32, khalf);
        wait_async_le0();
        __syncthreads();
        mm4(a, 35);
    }

    // epilogue: bias + tanh -> bf16
#pragma unroll
    for (int j = 0; j < 4; ++j) {
        int col = (nt0 + j) * 16 + (lane & 15);
        float bb = bias[col];
#pragma unroll
        for (int i = 0; i < 8; ++i) {
            int r = blockIdx.x * 64 + wave * 16 + i + 8 * khalf;
            hout[(size_t)r * Hc + col] = f2bf(tanhf(acc[j][i] + bb));
        }
    }
}

// Decoder step: C = hin @ [W_comb | W_out]  (K=1024, Ntiles=72)
// cols <1024 -> hout = tanh(C + b_comb) ; cols >=1024 -> y[b,t,:] = C + b_out
__global__ __launch_bounds__(128)
void dec_step_kernel(const u16t* __restrict__ hin,    // [B*H] bf16
                     const u16t* __restrict__ wpk,    // packed, Ntiles=72
                     const float* __restrict__ bcomb, // [H]
                     const float* __restrict__ bout,  // [D]
                     u16t* __restrict__ hout,         // [B*H] bf16
                     float* __restrict__ y,           // [B*T*D] fp32 (d_out)
                     int t) {
    __shared__ u16t smem[2 * 2048];
    const int tid   = threadIdx.x;
    const int lane  = tid & 31;
    const int wave  = tid >> 5;
    const int khalf = lane >> 4;
    const int row   = blockIdx.x * 64 + wave * 16 + (lane & 15);
    const int nt0   = blockIdx.y * 4;

    const u16t* arow_h = hin + (size_t)row * Hc;

    auto stage = [&](int s) {
        const u16t* g = wpk + ((size_t)s * 72 + nt0) * 512 + (size_t)tid * 8;
        unsigned l0 =
            (unsigned)(size_t)(smem + (size_t)(s & 1) * 2048 + tid * 8);
        ASYNC_LOAD_B128(l0, g);
        ASYNC_LOAD_B128(l0 + 2048u, g + 1024);
    };

    v8f acc[4] = {};
    auto mm4 = [&](v16bf a, int s) {
        const u16t* sb = smem + (size_t)(s & 1) * 2048;
        v16bf b0 = load_b_lds(sb, 0, lane);
        v16bf b1 = load_b_lds(sb, 1, lane);
        v16bf b2 = load_b_lds(sb, 2, lane);
        v16bf b3 = load_b_lds(sb, 3, lane);
        sched_fence();  // keep all ds loads before the WMMA group
        acc[0] = __builtin_amdgcn_wmma_f32_16x16x32_bf16(
            false, a, false, b0, (short)0, acc[0], false, false);
        acc[1] = __builtin_amdgcn_wmma_f32_16x16x32_bf16(
            false, a, false, b1, (short)0, acc[1], false, false);
        acc[2] = __builtin_amdgcn_wmma_f32_16x16x32_bf16(
            false, a, false, b2, (short)0, acc[2], false, false);
        acc[3] = __builtin_amdgcn_wmma_f32_16x16x32_bf16(
            false, a, false, b3, (short)0, acc[3], false, false);
    };

    stage(0);
#pragma unroll 1
    for (int s = 0; s < 31; ++s) {
        stage(s + 1);
        v16bf a = load_a(arow_h, s * 32, khalf);
        wait_async_le2();
        __syncthreads();
        mm4(a, s);
        __syncthreads();
    }
    {
        v16bf a = load_a(arow_h, 31 * 32, khalf);
        wait_async_le0();
        __syncthreads();
        mm4(a, 31);
    }

#pragma unroll
    for (int j = 0; j < 4; ++j) {
        int col = (nt0 + j) * 16 + (lane & 15);
        if (col < Hc) {
            float bb = bcomb[col];
#pragma unroll
            for (int i = 0; i < 8; ++i) {
                int r = blockIdx.x * 64 + wave * 16 + i + 8 * khalf;
                hout[(size_t)r * Hc + col] = f2bf(tanhf(acc[j][i] + bb));
            }
        } else {
            int d = col - Hc;
            float bb = bout[d];
#pragma unroll
            for (int i = 0; i < 8; ++i) {
                int r = blockIdx.x * 64 + wave * 16 + i + 8 * khalf;
                y[((size_t)r * Tc + t) * Dc + d] = acc[j][i] + bb;
            }
        }
    }
}

// ===========================================================================
// Host-side orchestration
// ===========================================================================
extern "C" void kernel_launch(void* const* d_in, const int* in_sizes, int n_in,
                              void* d_out, int out_size, void* d_ws,
                              size_t ws_size, hipStream_t stream) {
    const float* input     = (const float*)d_in[0];
    const float* eps       = (const float*)d_in[1];
    const float* W_enc_in  = (const float*)d_in[2];
    const float* W_enc_rec = (const float*)d_in[3];
    const float* b_enc     = (const float*)d_in[4];
    const float* W_mu      = (const float*)d_in[5];
    const float* b_mu      = (const float*)d_in[6];
    const float* W_std     = (const float*)d_in[7];
    const float* b_std     = (const float*)d_in[8];
    const float* W_h0      = (const float*)d_in[9];
    const float* b_h0      = (const float*)d_in[10];
    const float* W_out     = (const float*)d_in[11];
    const float* b_out     = (const float*)d_in[12];
    const float* W_dec_in  = (const float*)d_in[13];
    const float* W_dec_rec = (const float*)d_in[14];
    const float* b_dec     = (const float*)d_in[15];
    float* out = (float*)d_out;

    // ---- workspace carve (aligned 256B) ----
    char* wsp = (char*)d_ws;
    size_t off = 0;
    auto alloc = [&](size_t bytes) -> void* {
        void* p = wsp + off;
        off = (off + bytes + 255) & ~(size_t)255;
        return p;
    };
    u16t* x_bf    = (u16t*)alloc((size_t)Bc * Tc * Dc * 2);   // 67 MB
    u16t* encPack = (u16t*)alloc((size_t)36 * 64 * 512 * 2);  // 2.25 MB
    u16t* decPack = (u16t*)alloc((size_t)32 * 72 * 512 * 2);  // 2.25 MB
    u16t* henc0   = (u16t*)alloc((size_t)Bc * Hc * 2);
    u16t* henc1   = (u16t*)alloc((size_t)Bc * Hc * 2);
    u16t* hdec0   = (u16t*)alloc((size_t)Bc * Hc * 2);
    u16t* hdec1   = (u16t*)alloc((size_t)Bc * Hc * 2);
    float* Wcomb  = (float*)alloc((size_t)Hc * Hc * 4);       // 4 MB
    float* bcomb  = (float*)alloc((size_t)Hc * 4);
    float* zlat   = (float*)alloc((size_t)Bc * Lc * 4);
    (void)ws_size; (void)in_sizes; (void)n_in; (void)out_size;

    // ---- one-time per-call preparation ----
    {
        int n = Bc * Tc * Dc;
        cvt_bf16_kernel<<<(n + 255) / 256, 256, 0, stream>>>(input, x_bf, n);
    }
    {   // pack encoder stacked weight [W_enc_in ; W_enc_rec] (K=1152, N=1024)
        int tot_in = (Dc / 32) * (Hc / 16) * 512;
        pack_b_kernel<<<(tot_in + 255) / 256, 256, 0, stream>>>(
            W_enc_in, Dc, Hc, Hc, encPack, 64, 0, 0);
        int tot_rec = (Hc / 32) * (Hc / 16) * 512;
        pack_b_kernel<<<(tot_rec + 255) / 256, 256, 0, stream>>>(
            W_enc_rec, Hc, Hc, Hc, encPack, 64, 4, 0);
    }
    {   // decoder fused weight: W_comb then pack [W_comb | W_out] (N=1152)
        combine_w_kernel<<<(Hc * Hc + 255) / 256, 256, 0, stream>>>(
            W_out, W_dec_in, W_dec_rec, Wcomb);
        combine_b_kernel<<<(Hc + 255) / 256, 256, 0, stream>>>(
            b_out, W_dec_in, b_dec, bcomb);
        int tot_c = (Hc / 32) * (Hc / 16) * 512;
        pack_b_kernel<<<(tot_c + 255) / 256, 256, 0, stream>>>(
            Wcomb, Hc, Hc, Hc, decPack, 72, 0, 0);
        int tot_o = (Hc / 32) * (Dc / 16) * 512;
        pack_b_kernel<<<(tot_o + 255) / 256, 256, 0, stream>>>(
            W_out, Hc, Dc, Dc, decPack, 72, 0, 64);
    }
    {   // h_enc(0) = 0
        int n = Bc * Hc;
        zero_u16_kernel<<<(n + 255) / 256, 256, 0, stream>>>(henc0, n);
    }

    // ---- encoder recurrence: 512 sequential WMMA GEMM steps ----
    u16t* hping[2] = {henc0, henc1};
    for (int t = 0; t < Tc; ++t) {
        enc_step_kernel<<<dim3(Bc / 64, Hc / 64), 128, 0, stream>>>(
            x_bf, hping[t & 1], encPack, b_enc, hping[(t + 1) & 1], t);
    }
    u16t* hlast = hping[Tc & 1];  // T even -> buffer 0

    // ---- latent: mu, std, z ; decoder initial state ----
    latent_kernel<<<(Bc * Lc + 255) / 256, 256, 0, stream>>>(
        hlast, W_mu, b_mu, W_std, b_std, eps, out, zlat);
    h0_kernel<<<(Bc * Hc + 255) / 256, 256, 0, stream>>>(
        zlat, W_h0, b_h0, hdec0);

    // ---- decoder recurrence: 512 sequential fused WMMA GEMM steps ----
    u16t* dping[2] = {hdec0, hdec1};
    for (int t = 0; t < Tc; ++t) {
        dec_step_kernel<<<dim3(Bc / 64, 1152 / 64), 128, 0, stream>>>(
            dping[t & 1], decPack, bcomb, b_out, dping[(t + 1) & 1], out, t);
    }
}
